// BailingMoE_721554506403
// MI455X (gfx1250) — compile-verified, matrix operand order checked by reading, not run on hardware
//
#include <hip/hip_runtime.h>

typedef __bf16 bf16;
typedef __attribute__((ext_vector_type(16))) __bf16 v16bf;
typedef __attribute__((ext_vector_type(8)))  __bf16 bf16x8;
typedef __attribute__((ext_vector_type(4)))  __bf16 bf16x4;
typedef __attribute__((ext_vector_type(8)))  float  v8f;

#define T_TOK 1024
#define HDIM  2048
#define NEXP  16
#define IDIM  1408
#define TOPK  4

// ---------------------------------------------------------------------------
// init: zero per-expert counters
// ---------------------------------------------------------------------------
__global__ void moe_init_kern(int* __restrict__ ecnt) {
    if (threadIdx.x < NEXP) ecnt[threadIdx.x] = 0;
}

// ---------------------------------------------------------------------------
// router: logits -> softmax -> top-4 -> renorm -> expert grouping
// one thread per token
// ---------------------------------------------------------------------------
__global__ __launch_bounds__(256)
void moe_router_kern(const float* __restrict__ x, const float* __restrict__ gw,
                     int* __restrict__ ecnt, int* __restrict__ elist,
                     float* __restrict__ wgt) {
    int t = blockIdx.x * 256 + threadIdx.x;
    if (t >= T_TOK) return;
    const float4* xv = (const float4*)(x + (size_t)t * HDIM);
    float lg[NEXP];
    for (int e = 0; e < NEXP; ++e) {
        const float4* gv = (const float4*)(gw + (size_t)e * HDIM);
        float a0 = 0.f, a1 = 0.f, a2 = 0.f, a3 = 0.f;
        for (int j = 0; j < HDIM / 4; ++j) {
            float4 xa = xv[j], ga = gv[j];
            a0 += xa.x * ga.x; a1 += xa.y * ga.y;
            a2 += xa.z * ga.z; a3 += xa.w * ga.w;
        }
        lg[e] = (a0 + a1) + (a2 + a3);
    }
    float mx = lg[0];
    for (int e = 1; e < NEXP; ++e) mx = fmaxf(mx, lg[e]);
    float p[NEXP], s = 0.f;
    for (int e = 0; e < NEXP; ++e) { p[e] = __expf(lg[e] - mx); s += p[e]; }
    float inv = 1.f / s;
    for (int e = 0; e < NEXP; ++e) p[e] *= inv;
    int   idx[TOPK];
    float w4[TOPK];
    float ws = 0.f;
    for (int k = 0; k < TOPK; ++k) {
        int best = 0; float bv = -1.f;
        for (int e = 0; e < NEXP; ++e)
            if (p[e] > bv) { bv = p[e]; best = e; }
        idx[k] = best; w4[k] = bv; p[best] = -1.f; ws += bv;
    }
    float wn = 1.f / ws;
    for (int k = 0; k < TOPK; ++k) {
        int e = idx[k];
        int pos = atomicAdd(&ecnt[e], 1);
        int code = (t << 2) | k;
        elist[(e << 10) + pos] = code;
        wgt[code] = w4[k] * wn;
    }
}

// ---------------------------------------------------------------------------
// Unified WMMA GEMM:
//   GATEUP : computes g = A@Wg^T, u = A@Wu^T (Wu rows at uRowOff), stores
//            bf16 silu(g)*u
//   GATHER : rows taken from per-expert slot list (code = t*4+k);
//            A row = code >> GSHIFT, out row = code; early exit on count;
//            invalid rows clamp to row 0 (results discarded in epilogue)
//   ABF16  : A operand already bf16 -> staged with global_load_async_to_lds
//   SCALE  : multiply output row by wgt[code]
// Block: 256 threads (8 waves), tile 128x64, BK=32, wave tile 32x32 (2x2 WMMA)
// ---------------------------------------------------------------------------
template <bool GATEUP, bool GATHER, bool ABF16, bool SCALE, int GSHIFT>
__global__ __launch_bounds__(256)
void moe_gemm_kern(const void* __restrict__ Ap, const float* __restrict__ Bp,
                   void* __restrict__ Outp,
                   const int* __restrict__ elist, const int* __restrict__ ecnt,
                   const float* __restrict__ wgt,
                   int K, int Astride, int Bstride, int OutStride,
                   int uRowOff, long long eStrideB, int Mtotal) {
    constexpr int BM = 128, BN = 64, BK = 32, LDSK = BK + 8;
    constexpr int BROWS = GATEUP ? 2 * BN : BN;
    constexpr int AES = ABF16 ? 2 : 4;           // A element size bytes
    constexpr int AR = ABF16 ? 2 : 4;            // A staging rounds
    constexpr int BR = BROWS / 32;               // B staging rounds

    __shared__ __align__(16) bf16 As[BM * LDSK];
    __shared__ __align__(16) bf16 Bs[BROWS * LDSK];
    __shared__ int rowcode[BM];

    const int tid   = threadIdx.x;
    const int e     = blockIdx.z;
    const int rows  = GATHER ? ecnt[e] : Mtotal;
    const int mBase = blockIdx.y * BM;
    if (mBase >= rows) return;                   // uniform early exit
    const int nBase = blockIdx.x * BN;
    const float* Bbase = Bp + (long long)e * eStrideB;
    const int* el = GATHER ? (elist + (e << 10)) : nullptr;

    if (tid < BM) {
        int s = mBase + tid;
        rowcode[tid] = GATHER ? ((s < rows) ? el[s] : -1) : s;
    }
    __syncthreads();

    // ---- precompute staging pointers (fixed across k-loop, branchless) ----
    const char* aB[AR]; int aoff[AR];
    #pragma unroll
    for (int r = 0; r < AR; ++r) {
        int idx = r * 256 + tid;
        int row, col;
        if (ABF16) { row = idx >> 2; col = (idx & 3) * 8; }
        else       { row = idx >> 3; col = (idx & 7) * 4; }
        int code = rowcode[row];
        // invalid rows clamp to A row 0: loads stay valid, results discarded
        long long grow = (code >= 0)
                       ? (GATHER ? (long long)(code >> GSHIFT) : (long long)code)
                       : 0ll;
        aB[r]   = (const char*)Ap + (grow * Astride + col) * AES;
        aoff[r] = row * LDSK + col;
    }
    const float* bP[BR]; int boff[BR];
    #pragma unroll
    for (int r = 0; r < BR; ++r) {
        int idx = r * 256 + tid;
        int row = idx >> 3;
        int col = (idx & 7) * 4;
        int g = (row < BN) ? (nBase + row) : (uRowOff + nBase + (row - BN));
        bP[r] = Bbase + (long long)g * Bstride + col;
        boff[r] = row * LDSK + col;
    }

    const int lane = tid & 31, wv = tid >> 5;
    const int wm = (wv & 3) * 32, wn = (wv >> 2) * 32;
    const int h = lane >> 4, lr = lane & 15;

    v8f accg[2][2] = {};
    v8f accu[2][2] = {};

    for (int k0 = 0; k0 < K; k0 += BK) {
        // prefetch next tiles (global_prefetch_b8)
        if (k0 + BK < K) {
            __builtin_prefetch(aB[0] + (size_t)(k0 + BK) * AES, 0, 3);
            __builtin_prefetch((const char*)bP[0] + (size_t)(k0 + BK) * 4, 0, 3);
        }
        // ---- stage A ----
        if (ABF16) {
            // bf16 activations: DMA straight into LDS, no VGPR round-trip
            #pragma unroll
            for (int r = 0; r < AR; ++r) {
                unsigned ldst = (unsigned)(size_t)(&As[aoff[r]]);
                const void* gsrc = aB[r] + (size_t)k0 * 2;
                asm volatile("global_load_async_to_lds_b128 %0, %1, off"
                             :: "v"(ldst), "v"(gsrc) : "memory");
            }
        } else {
            #pragma unroll
            for (int r = 0; r < AR; ++r) {
                float4 v = *(const float4*)(aB[r] + (size_t)k0 * 4);
                bf16x4 o;
                o[0] = (bf16)v.x; o[1] = (bf16)v.y;
                o[2] = (bf16)v.z; o[3] = (bf16)v.w;
                *(bf16x4*)&As[aoff[r]] = o;
            }
        }
        // ---- stage B (fp32 weights -> bf16 in registers) ----
        #pragma unroll
        for (int r = 0; r < BR; ++r) {
            float4 v = *(const float4*)((const char*)bP[r] + (size_t)k0 * 4);
            bf16x4 o;
            o[0] = (bf16)v.x; o[1] = (bf16)v.y;
            o[2] = (bf16)v.z; o[3] = (bf16)v.w;
            *(bf16x4*)&Bs[boff[r]] = o;
        }
        if (ABF16) {
            asm volatile("s_wait_asynccnt 0x0" ::: "memory");
        }
        __syncthreads();

        // ---- fragments (per-lane K layout from ISA 16-bit A table) ----
        v16bf af[2], bfz[2], buz[2];
        #pragma unroll
        for (int i = 0; i < 2; ++i) {
            const bf16* p = &As[(wm + i * 16 + lr) * LDSK];
            bf16x8 lo = *(const bf16x8*)(p + h * 8);
            bf16x8 hi = *(const bf16x8*)(p + 16 + h * 8);
            af[i] = __builtin_shufflevector(lo, hi, 0,1,2,3,4,5,6,7,8,9,10,11,12,13,14,15);
        }
        #pragma unroll
        for (int j = 0; j < 2; ++j) {
            const bf16* p = &Bs[(wn + j * 16 + lr) * LDSK];
            bf16x8 lo = *(const bf16x8*)(p + h * 8);
            bf16x8 hi = *(const bf16x8*)(p + 16 + h * 8);
            bfz[j] = __builtin_shufflevector(lo, hi, 0,1,2,3,4,5,6,7,8,9,10,11,12,13,14,15);
            if (GATEUP) {
                const bf16* q = &Bs[(BN + wn + j * 16 + lr) * LDSK];
                bf16x8 ulo = *(const bf16x8*)(q + h * 8);
                bf16x8 uhi = *(const bf16x8*)(q + 16 + h * 8);
                buz[j] = __builtin_shufflevector(ulo, uhi, 0,1,2,3,4,5,6,7,8,9,10,11,12,13,14,15);
            }
        }
        #pragma unroll
        for (int i = 0; i < 2; ++i)
            #pragma unroll
            for (int j = 0; j < 2; ++j) {
                accg[i][j] = __builtin_amdgcn_wmma_f32_16x16x32_bf16(
                    false, af[i], false, bfz[j], (short)0, accg[i][j], false, false);
                if (GATEUP)
                    accu[i][j] = __builtin_amdgcn_wmma_f32_16x16x32_bf16(
                        false, af[i], false, buz[j], (short)0, accu[i][j], false, false);
            }
        __syncthreads();
    }

    // ---- epilogue ----
    #pragma unroll
    for (int i = 0; i < 2; ++i) {
        #pragma unroll
        for (int v = 0; v < 8; ++v) {
            int mloc = wm + i * 16 + h * 8 + v;   // C layout: VGPR v, lane-half h
            int code = rowcode[mloc];
            if (code < 0) continue;
            float rs = SCALE ? wgt[code] : 1.f;
            #pragma unroll
            for (int j = 0; j < 2; ++j) {
                int col = nBase + wn + j * 16 + lr;
                if (GATEUP) {
                    float g = accg[i][j][v];
                    float u = accu[i][j][v];
                    float a = (g / (1.f + __expf(-g))) * u;  // silu(g)*u
                    ((bf16*)Outp)[(long long)code * OutStride + col] = (bf16)a;
                } else {
                    float val = accg[i][j][v] * rs;
                    ((float*)Outp)[(long long)code * OutStride + col] = val;
                }
            }
        }
    }
}

// ---------------------------------------------------------------------------
// combine: out[t,h] (= shared_out) += sum_k slot_out[t*4+k, h]
// ---------------------------------------------------------------------------
__global__ __launch_bounds__(256)
void moe_combine_kern(float* __restrict__ out, const float* __restrict__ slot) {
    int i = blockIdx.x * 256 + threadIdx.x;       // over T*H/4
    int hc = i & (HDIM / 4 - 1);
    int t = i >> 9;
    float4 o = ((const float4*)out)[i];
    #pragma unroll
    for (int k = 0; k < TOPK; ++k) {
        float4 s = ((const float4*)(slot + ((size_t)((t << 2) | k)) * HDIM))[hc];
        o.x += s.x; o.y += s.y; o.z += s.z; o.w += s.w;
    }
    ((float4*)out)[i] = o;
}

// ---------------------------------------------------------------------------
extern "C" void kernel_launch(void* const* d_in, const int* in_sizes, int n_in,
                              void* d_out, int out_size, void* d_ws, size_t ws_size,
                              hipStream_t stream) {
    const float* x   = (const float*)d_in[0];   // [T,H]
    const float* gw  = (const float*)d_in[1];   // [E,H]
    const float* wgu = (const float*)d_in[2];   // [E,2I,H]
    const float* wdn = (const float*)d_in[3];   // [E,H,I]
    const float* sgu = (const float*)d_in[4];   // [2*I*S, H] = [5632,2048]
    const float* sdn = (const float*)d_in[5];   // [H, I*S]  = [2048,2816]
    float* out = (float*)d_out;

    char* w = (char*)d_ws;
    int*   ecnt  = (int*)w;                                     // 256 B
    int*   elist = (int*)(w + 256);                             // 16*1024*4
    float* wgt   = (float*)(w + 256 + 65536);                   // 4096*4
    bf16*  act_s = (bf16*)(w + 82176);                          // 1024*2816 bf16
    bf16*  act_r = (bf16*)(w + 82176 + 5767168);                // 4096*1408 bf16
    float* slot  = (float*)(w + 82176 + 5767168 + 11534336);    // 4096*2048 f32

    moe_init_kern<<<1, 32, 0, stream>>>(ecnt);
    moe_router_kern<<<4, 256, 0, stream>>>(x, gw, ecnt, elist, wgt);

    // routed GEMM1: x(gathered) @ w_gate_up[e]^T, silu*mul -> act_r (bf16)
    moe_gemm_kern<true, true, false, false, 2>
        <<<dim3(IDIM / 64, T_TOK / 128, NEXP), 256, 0, stream>>>(
            x, wgu, act_r, elist, ecnt, nullptr,
            HDIM, HDIM, HDIM, IDIM, IDIM, (long long)2 * IDIM * HDIM, T_TOK);

    // routed GEMM2: act_r(gathered) @ w_down[e]^T, *wgt -> slot_out (f32)
    moe_gemm_kern<false, true, true, true, 0>
        <<<dim3(HDIM / 64, T_TOK / 128, NEXP), 256, 0, stream>>>(
            act_r, wdn, slot, elist, ecnt, wgt,
            IDIM, IDIM, IDIM, HDIM, 0, (long long)HDIM * IDIM, T_TOK);

    // shared GEMM1: x @ shared_gate_up^T, silu*mul -> act_s (bf16, width 2816)
    moe_gemm_kern<true, false, false, false, 0>
        <<<dim3(2816 / 64, T_TOK / 128, 1), 256, 0, stream>>>(
            x, sgu, act_s, nullptr, nullptr, nullptr,
            HDIM, HDIM, HDIM, 2816, 2816, 0, T_TOK);

    // shared GEMM2: act_s @ shared_down^T -> d_out (f32)
    moe_gemm_kern<false, false, true, false, 0>
        <<<dim3(HDIM / 64, T_TOK / 128, 1), 256, 0, stream>>>(
            act_s, sdn, out, nullptr, nullptr, nullptr,
            2816, 2816, 2816, HDIM, 0, 0, T_TOK);

    moe_combine_kern<<<(T_TOK * HDIM / 4) / 256, 256, 0, stream>>>(out, slot);
}
